// RNNHyperTokenDecoder_44409961840881
// MI455X (gfx1250) — compile-verified, matrix operand order checked by reading, not run on hardware
//
#include <hip/hip_runtime.h>
#include <hip/hip_bf16.h>
#include <stdint.h>

// Model dims
#define E_   512
#define H_   1024
#define T_   128
#define V_   32000
#define HT_  512
#define B_   32

typedef __attribute__((ext_vector_type(16))) __bf16 v16bf;
typedef __attribute__((ext_vector_type(8)))  float  v8f;
typedef __attribute__((ext_vector_type(4)))  unsigned int u32x4;
typedef __attribute__((ext_vector_type(4)))  unsigned int u32x4v;
typedef __attribute__((ext_vector_type(8)))  int i32x8v;
typedef __attribute__((ext_vector_type(4)))  int i32x4v;

union FragA { v16bf v; u32x4 u[2]; };
union FragC { v8f  v; float f[8]; };

#if __has_builtin(__builtin_amdgcn_tensor_load_to_lds)
#define USE_TDM 1
#define ASYNC_PENDING "0x2"
#else
#define USE_TDM 0
#define ASYNC_PENDING "0x4"
#endif

// ---------------------------------------------------------------------------
// WMMA bf16 fragment mapping (CDNA5 ISA, 16-bit A 16x32):
//   lane l: row = l&15, half = l>>4
//   j in [0,16): k = (j&7) + ((j>>3)<<4) + (half<<3)
// Packed B layout: P[(((nb*Kb + kb)*32 + lane)*16 + j)]  (1 KB per fragment)
// ---------------------------------------------------------------------------

__global__ void pack_w_kernel(const float* __restrict__ W, __bf16* __restrict__ P,
                              int N, int K) {
  int idx = blockIdx.x * blockDim.x + threadIdx.x;
  if (idx >= N * K) return;
  const int Kb   = K >> 5;
  const int j    = idx & 15;
  const int lane = (idx >> 4) & 31;
  const int rem  = idx >> 9;
  const int kb   = rem % Kb;
  const int nb   = rem / Kb;
  const int row  = nb * 16 + (lane & 15);
  const int half = lane >> 4;
  const int k    = kb * 32 + (j & 7) + ((j >> 3) << 4) + (half << 3);
  P[idx] = (__bf16)W[(size_t)row * K + k];
}

__global__ void cvt_bf16_kernel(const float* __restrict__ X, __bf16* __restrict__ Y, int n) {
  int i = blockIdx.x * blockDim.x + threadIdx.x;
  if (i < n) Y[i] = (__bf16)X[i];
}

__global__ void add_vec_kernel(const float* __restrict__ a, const float* __restrict__ b,
                               float* __restrict__ out, int n) {
  int i = blockIdx.x * blockDim.x + threadIdx.x;
  if (i < n) out[i] = a[i] + b[i];
}

// seq[t*B + b][h] = expanded[b][h] + pos[t][h]  (bf16)
__global__ void build_seq_kernel(const float* __restrict__ exp_f32,
                                 const float* __restrict__ pos,
                                 __bf16* __restrict__ seq) {
  int idx = blockIdx.x * blockDim.x + threadIdx.x;
  if (idx >= T_ * B_ * H_) return;
  const int h = idx & (H_ - 1);
  const int r = idx >> 10;
  const int b = r & (B_ - 1);
  const int t = r >> 5;
  seq[idx] = (__bf16)(exp_f32[b * H_ + h] + pos[t * H_ + h]);
}

#if USE_TDM
// One TDM 2D descriptor: 8 rows (packed fragments, stride Kb*512 elems) x 512
// bf16 each -> 8 KB tile into LDS.  Issued once per block (wave 0).
__device__ __forceinline__ void tdm_load_b_tile(const __bf16* gptr, uint32_t lds_off,
                                                unsigned tensor_d0, unsigned tensor_d1,
                                                unsigned stride_elems) {
  const uint64_t ga = (uint64_t)(uintptr_t)gptr;
  u32x4v g0;
  g0.x = 1u;                                                 // count=1 descriptor
  g0.y = lds_off;                                            // lds_addr (bytes)
  g0.z = (uint32_t)ga;                                       // global_addr[31:0]
  g0.w = (uint32_t)((ga >> 32) & 0x1FFFFFFu) | (2u << 30);   // addr[56:32] | type=2
  i32x8v g1;
  g1[0] = (int)(1u << 16);                                   // data_size=1 (2B), mask=0
  g1[1] = (int)((tensor_d0 & 0xFFFFu) << 16);                // tensor_dim0 -> bits 79:48
  g1[2] = (int)((tensor_d0 >> 16) | ((tensor_d1 & 0xFFFFu) << 16));
  g1[3] = (int)((tensor_d1 >> 16) | (512u << 16));           // tile_dim0 = 512
  g1[4] = (int)8;                                            // tile_dim1 = 8, tile_dim2 = 0
  g1[5] = (int)stride_elems;                                 // tensor_dim0_stride[31:0]
  g1[6] = 0;
  g1[7] = 0;
  i32x4v gz = {0, 0, 0, 0};
#if __clang_major__ >= 23
  i32x8v gz8 = {0, 0, 0, 0, 0, 0, 0, 0};
  __builtin_amdgcn_tensor_load_to_lds(g0, g1, gz, gz, gz8, 0);
#else
  __builtin_amdgcn_tensor_load_to_lds(g0, g1, gz, gz, 0);
#endif
}
#endif

// ---------------------------------------------------------------------------
// bf16 WMMA GEMM with async LDS staging:
//   Out[M][N] = A[M][K] @ packedB(N,K) + bias[N]
// 256 threads = 8 waves (4 M x 2 N); wave tile 32x64 (2x4 WMMA tiles).
// Per kb: 8 KB A tile (128x32) via global_load_async_to_lds_b128 (ASYNCcnt),
//         8 KB B tile (8 fragments) via TDM tensor_load_to_lds (TENSORcnt),
// double-buffered; steady-state loop body is branch-free (last iter peeled).
// remap=1: output row m = t*B+b scattered to b*T+t (final logits [B,T,V]).
// ---------------------------------------------------------------------------
__global__ __launch_bounds__(256)
void wmma_gemm_kernel(const __bf16* __restrict__ A, const __bf16* __restrict__ Bp,
                      const float* __restrict__ bias, float* __restrict__ Out,
                      int M, int N, int K, int remap) {
  __shared__ __bf16 smemA[2 * 4096];   // 2 x 128x32 bf16
  __shared__ __bf16 smemB[2 * 4096];   // 2 x 8x512  bf16
  const int tid  = threadIdx.x;
  const int lane = tid & 31;
  const int wave = tid >> 5;
  const int wm   = wave & 3;
  const int wn   = wave >> 2;
  const int Kb   = K >> 5;
  const int row0 = lane & 15;
  const int half = lane >> 4;
  const int nb0  = blockIdx.y * 8;

  // A staging: thread copies 16B chunks tid and tid+256 of the 8 KB tile.
  int arow0 = blockIdx.x * 128 + (tid >> 2);
  if (arow0 >= M) arow0 = M - 1;                 // clamp (stores are guarded)
  int arow1 = blockIdx.x * 128 + ((tid + 256) >> 2);
  if (arow1 >= M) arow1 = M - 1;
  const int acol = (tid & 3) * 8;                // (tid+256)&3 == tid&3
  const uint32_t ldsA  = (uint32_t)(uintptr_t)smemA;
  const uint32_t ldsB  = (uint32_t)(uintptr_t)smemB;
  const uint32_t lofs0 = (uint32_t)tid * 16u;
  const uint32_t lofs1 = (uint32_t)(tid + 256) * 16u;

  auto issue_tile = [&](int kb, int bsel) {
    const __bf16* ga0 = A + (size_t)arow0 * K + kb * 32 + acol;
    const __bf16* ga1 = A + (size_t)arow1 * K + kb * 32 + acol;
    uint32_t l0 = ldsA + (uint32_t)bsel * 8192u + lofs0;
    uint32_t l1 = ldsA + (uint32_t)bsel * 8192u + lofs1;
    asm volatile("global_load_async_to_lds_b128 %0, %1, off"
                 :: "v"(l0), "v"(ga0) : "memory");
    asm volatile("global_load_async_to_lds_b128 %0, %1, off"
                 :: "v"(l1), "v"(ga1) : "memory");
#if USE_TDM
    if (wave == 0)
      tdm_load_b_tile(Bp + ((size_t)nb0 * Kb + kb) * 512,
                      ldsB + (uint32_t)bsel * 8192u,
                      (unsigned)(Kb * 512), (unsigned)(N >> 4),
                      (unsigned)(Kb * 512));
#else
    // Fallback: B tile also via async b128 copies (rows have stride Kb*1KB).
    {
      int brow0 = tid >> 6,         bcol0 = (tid & 63) * 8;
      int brow1 = (tid + 256) >> 6, bcol1 = (tid & 63) * 8;
      const __bf16* gb0 = Bp + ((size_t)(nb0 + brow0) * Kb + kb) * 512 + bcol0;
      const __bf16* gb1 = Bp + ((size_t)(nb0 + brow1) * Kb + kb) * 512 + bcol1;
      uint32_t b0 = ldsB + (uint32_t)bsel * 8192u + lofs0;
      uint32_t b1 = ldsB + (uint32_t)bsel * 8192u + lofs1;
      asm volatile("global_load_async_to_lds_b128 %0, %1, off"
                   :: "v"(b0), "v"(gb0) : "memory");
      asm volatile("global_load_async_to_lds_b128 %0, %1, off"
                   :: "v"(b1), "v"(gb1) : "memory");
    }
#endif
  };

  FragC acc[2][4];
#pragma unroll
  for (int i = 0; i < 2; ++i)
#pragma unroll
    for (int j = 0; j < 4; ++j)
#pragma unroll
      for (int r = 0; r < 8; ++r) acc[i][j].f[r] = 0.0f;

  auto compute_tile = [&](int cur) {
    const __bf16* At = smemA + cur * 4096;
    const __bf16* Bt = smemB + cur * 4096;
    FragA a[2];
#pragma unroll
    for (int i = 0; i < 2; ++i) {
      const __bf16* ap = At + (wm * 32 + i * 16 + row0) * 32 + half * 8;
      a[i].u[0] = *(const u32x4*)ap;        // ds_load_b128
      a[i].u[1] = *(const u32x4*)(ap + 16);
    }
#pragma unroll
    for (int j = 0; j < 4; ++j) {
      const __bf16* bp = Bt + (wn * 4 + j) * 512 + lane * 16;
      FragA b;
      b.u[0] = *(const u32x4*)bp;
      b.u[1] = *(const u32x4*)(bp + 8);
#pragma unroll
      for (int i = 0; i < 2; ++i)
        acc[i][j].v = __builtin_amdgcn_wmma_f32_16x16x32_bf16(
            false, a[i].v, false, b.v, (short)0, acc[i][j].v, false, false);
    }
  };

  issue_tile(0, 0);

  // Steady state: branch-free pipelined body.
  for (int kb = 0; kb + 1 < Kb; ++kb) {
    const int cur = kb & 1;
    issue_tile(kb + 1, cur ^ 1);   // buf^1 readers finished at barrier of kb-1
    asm volatile("s_wait_asynccnt " ASYNC_PENDING ::: "memory");
#if USE_TDM
    __builtin_amdgcn_s_wait_tensorcnt(1);   // free for waves with TENSORcnt==0
#endif
    __syncthreads();               // tile kb visible to all 8 waves
    compute_tile(cur);
    __syncthreads();               // all reads of buf cur done before refill
  }

  // Last tile.
  asm volatile("s_wait_asynccnt 0x0" ::: "memory");
#if USE_TDM
  __builtin_amdgcn_s_wait_tensorcnt(0);
#endif
  __syncthreads();
  compute_tile((Kb - 1) & 1);

  const int mBase = blockIdx.x * 128 + wm * 32;
  const int nBase = blockIdx.y * 128 + wn * 64;
#pragma unroll
  for (int i = 0; i < 2; ++i)
#pragma unroll
    for (int j = 0; j < 4; ++j) {
      const int n  = nBase + j * 16 + row0;
      const float bv = bias[n];
#pragma unroll
      for (int r = 0; r < 8; ++r) {
        const int m = mBase + i * 16 + half * 8 + r;
        if (m < M) {
          const size_t orow = remap ? (size_t)((m & (B_ - 1)) * T_ + (m >> 5))
                                    : (size_t)m;
          Out[orow * (size_t)N + n] = acc[i][j].f[r] + bv;
        }
      }
    }
}

// ---------------------------------------------------------------------------
// Row LayerNorm + ReLU; optional fp32 and bf16 outputs.
// ---------------------------------------------------------------------------
__global__ __launch_bounds__(256)
void ln_relu_kernel(const float* __restrict__ X, const float* __restrict__ g,
                    const float* __restrict__ be, float* __restrict__ Yf,
                    __bf16* __restrict__ Yb, int C) {
  const int row = blockIdx.x;
  const float* x = X + (size_t)row * C;
  __shared__ float redS[8], redS2[8];
  __shared__ float muS, rstdS;
  float s = 0.f, s2 = 0.f;
  for (int c = threadIdx.x; c < C; c += blockDim.x) { float v = x[c]; s += v; s2 += v * v; }
  for (int o = 16; o > 0; o >>= 1) { s += __shfl_xor(s, o); s2 += __shfl_xor(s2, o); }
  const int wave = threadIdx.x >> 5;
  if ((threadIdx.x & 31) == 0) { redS[wave] = s; redS2[wave] = s2; }
  __syncthreads();
  if (threadIdx.x == 0) {
    float S = 0.f, S2 = 0.f;
    const int nw = blockDim.x >> 5;
    for (int w = 0; w < nw; ++w) { S += redS[w]; S2 += redS2[w]; }
    const float mu = S / C;
    muS = mu;
    rstdS = rsqrtf(S2 / C - mu * mu + 1e-5f);
  }
  __syncthreads();
  const float mu = muS, rstd = rstdS;
  for (int c = threadIdx.x; c < C; c += blockDim.x) {
    float v = (x[c] - mu) * rstd * g[c] + be[c];
    v = fmaxf(v, 0.f);
    if (Yf) Yf[(size_t)row * C + c] = v;
    if (Yb) Yb[(size_t)row * C + c] = (__bf16)v;
  }
}

// ---------------------------------------------------------------------------
// Recurrent scan: one workgroup (32 waves) per direction (blockIdx.x = dir).
// h state (32x1024 bf16) double-buffered in LDS (128 KB of the 320 KB pool).
// Per step: h' = relu(Xpre[t] + h @ Whh.T).  Whh is L2-resident (2 MB/dir).
// ---------------------------------------------------------------------------
__global__ __launch_bounds__(1024)
void rnn_scan_kernel(const float* __restrict__ XpreBase,
                     const __bf16* __restrict__ WhhBase,
                     const __bf16* __restrict__ h0,
                     __bf16* __restrict__ Yout) {
  const int dir = blockIdx.x;
  const float*  Xpre = XpreBase + (size_t)dir * T_ * B_ * H_;
  const __bf16* Whh  = WhhBase + (size_t)dir * H_ * H_;
  extern __shared__ __bf16 hbuf[];  // 2 * 32 * 1024 bf16
  const int tid = threadIdx.x;
  for (int i = tid; i < B_ * H_; i += 1024) hbuf[i] = h0[i];
  __syncthreads();

  const int lane  = tid & 31;
  const int wave  = tid >> 5;
  const int row0  = lane & 15;
  const int half  = lane >> 4;
  const int nBase = wave * 32;

  for (int t = 0; t < T_; ++t) {
    const int tt = dir ? (T_ - 1 - t) : t;
    const __bf16* hcur  = hbuf + (size_t)(t & 1) * (B_ * H_);
    __bf16*       hnext = hbuf + (size_t)((t + 1) & 1) * (B_ * H_);

    FragC acc[2][2];
#pragma unroll
    for (int i = 0; i < 2; ++i)
#pragma unroll
      for (int j = 0; j < 2; ++j) {
        const int n = nBase + j * 16 + row0;
#pragma unroll
        for (int r = 0; r < 8; ++r) {
          const int m = i * 16 + half * 8 + r;
          acc[i][j].f[r] = Xpre[((size_t)tt * B_ + m) * H_ + n];
        }
      }

    for (int kb = 0; kb < 32; ++kb) {
      FragA a[2];
#pragma unroll
      for (int i = 0; i < 2; ++i) {
        const __bf16* ap = hcur + (i * 16 + row0) * H_ + kb * 32 + half * 8;
        a[i].u[0] = *(const u32x4*)ap;        // ds_load_b128
        a[i].u[1] = *(const u32x4*)(ap + 16);
      }
#pragma unroll
      for (int j = 0; j < 2; ++j) {
        const int nb = (nBase >> 4) + j;
        const __bf16* bp = Whh + ((((size_t)nb * 32 + kb) * 32 + lane) << 4);
        __builtin_prefetch(bp + 512, 0, 1);   // next kb fragment (unconditional)
        FragA b;
        b.u[0] = *(const u32x4*)bp;
        b.u[1] = *(const u32x4*)(bp + 8);
#pragma unroll
        for (int i = 0; i < 2; ++i)
          acc[i][j].v = __builtin_amdgcn_wmma_f32_16x16x32_bf16(
              false, a[i].v, false, b.v, (short)0, acc[i][j].v, false, false);
      }
    }

#pragma unroll
    for (int i = 0; i < 2; ++i)
#pragma unroll
      for (int j = 0; j < 2; ++j) {
        const int n = nBase + j * 16 + row0;
#pragma unroll
        for (int r = 0; r < 8; ++r) {
          const int m = i * 16 + half * 8 + r;
          float v = fmaxf(acc[i][j].f[r], 0.f);
          __bf16 bv = (__bf16)v;
          hnext[m * H_ + n] = bv;
          Yout[((size_t)tt * B_ + m) * (2 * H_) + dir * H_ + n] = bv;
        }
      }
    __syncthreads();
  }
}

// ---------------------------------------------------------------------------
extern "C" void kernel_launch(void* const* d_in, const int* in_sizes, int n_in,
                              void* d_out, int out_size, void* d_ws, size_t ws_size,
                              hipStream_t stream) {
  (void)in_sizes; (void)n_in; (void)out_size; (void)ws_size;
  const float* x     = (const float*)d_in[0];
  const float* W_exp = (const float*)d_in[1];
  const float* b_exp = (const float*)d_in[2];
  const float* g1    = (const float*)d_in[3];
  const float* be1   = (const float*)d_in[4];
  const float* pos   = (const float*)d_in[5];
  const float* Wih0  = (const float*)d_in[6];
  const float* Whh0  = (const float*)d_in[7];
  const float* bih0  = (const float*)d_in[8];
  const float* bhh0  = (const float*)d_in[9];
  const float* Wih1  = (const float*)d_in[10];
  const float* Whh1  = (const float*)d_in[11];
  const float* bih1  = (const float*)d_in[12];
  const float* bhh1  = (const float*)d_in[13];
  const float* Wp1   = (const float*)d_in[14];
  const float* bp1   = (const float*)d_in[15];
  const float* g2    = (const float*)d_in[16];
  const float* be2   = (const float*)d_in[17];
  const float* Wp2   = (const float*)d_in[18];
  const float* bp2   = (const float*)d_in[19];
  float* logits = (float*)d_out;

  uint8_t* ws = (uint8_t*)d_ws;
  size_t off = 0;
  auto alloc = [&](size_t bytes) -> void* {
    void* p = ws + off;
    off += (bytes + 255) & ~(size_t)255;
    return p;
  };
  __bf16* pW_exp  = (__bf16*)alloc((size_t)H_ * HT_ * 2);
  __bf16* pWih0   = (__bf16*)alloc((size_t)2 * H_ * H_ * 2);
  __bf16* pWhh0   = (__bf16*)alloc((size_t)2 * H_ * H_ * 2);
  __bf16* pWih1   = (__bf16*)alloc((size_t)2 * H_ * 2 * H_ * 2);
  __bf16* pWhh1   = (__bf16*)alloc((size_t)2 * H_ * H_ * 2);
  __bf16* pWp1    = (__bf16*)alloc((size_t)E_ * 2 * H_ * 2);
  __bf16* pWp2    = (__bf16*)alloc((size_t)V_ * E_ * 2);
  __bf16* x_bf    = (__bf16*)alloc((size_t)B_ * HT_ * 2);
  float*  exp_pre = (float*)alloc((size_t)B_ * H_ * 4);
  float*  exp_f   = (float*)alloc((size_t)B_ * H_ * 4);
  __bf16* exp_b   = (__bf16*)alloc((size_t)B_ * H_ * 2);
  float*  comb0   = (float*)alloc((size_t)2 * H_ * 4);
  float*  comb1   = (float*)alloc((size_t)2 * H_ * 4);
  __bf16* seq     = (__bf16*)alloc((size_t)T_ * B_ * H_ * 2);
  float*  Xpre    = (float*)alloc((size_t)2 * T_ * B_ * H_ * 4);  // reused L0/L1
  __bf16* y0      = (__bf16*)alloc((size_t)T_ * B_ * 2 * H_ * 2);
  __bf16* y1      = (__bf16*)alloc((size_t)T_ * B_ * 2 * H_ * 2);
  float*  hp_f    = (float*)alloc((size_t)T_ * B_ * E_ * 4);
  __bf16* hp_b    = (__bf16*)alloc((size_t)T_ * B_ * E_ * 2);

  auto packW = [&](const float* W, __bf16* P, int N, int K) {
    int total = N * K;
    pack_w_kernel<<<(total + 255) / 256, 256, 0, stream>>>(W, P, N, K);
  };
  packW(W_exp, pW_exp, H_, HT_);
  packW(Wih0,               pWih0,               H_, H_);
  packW(Wih0 + H_ * H_,     pWih0 + H_ * H_,     H_, H_);
  packW(Whh0,               pWhh0,               H_, H_);
  packW(Whh0 + H_ * H_,     pWhh0 + H_ * H_,     H_, H_);
  packW(Wih1,                 pWih1,                 H_, 2 * H_);
  packW(Wih1 + H_ * 2 * H_,   pWih1 + H_ * 2 * H_,   H_, 2 * H_);
  packW(Whh1,               pWhh1,               H_, H_);
  packW(Whh1 + H_ * H_,     pWhh1 + H_ * H_,     H_, H_);
  packW(Wp1, pWp1, E_, 2 * H_);
  packW(Wp2, pWp2, V_, E_);

  cvt_bf16_kernel<<<(B_ * HT_ + 255) / 256, 256, 0, stream>>>(x, x_bf, B_ * HT_);
  add_vec_kernel<<<(2 * H_ + 255) / 256, 256, 0, stream>>>(bih0, bhh0, comb0, 2 * H_);
  add_vec_kernel<<<(2 * H_ + 255) / 256, 256, 0, stream>>>(bih1, bhh1, comb1, 2 * H_);

  // expand: [32,512] @ [512,1024] + b_exp, then LN+ReLU
  {
    dim3 g(1, H_ / 128);
    wmma_gemm_kernel<<<g, 256, 0, stream>>>(x_bf, pW_exp, b_exp, exp_pre, B_, H_, HT_, 0);
  }
  ln_relu_kernel<<<B_, 256, 0, stream>>>(exp_pre, g1, be1, exp_f, exp_b, H_);

  build_seq_kernel<<<(T_ * B_ * H_ + 255) / 256, 256, 0, stream>>>(exp_f, pos, seq);

  // Layer 0: hoisted input GEMM (both dirs), then recurrent scan
  {
    dim3 g(T_ * B_ / 128, H_ / 128);
    wmma_gemm_kernel<<<g, 256, 0, stream>>>(seq, pWih0, comb0, Xpre,
                                            T_ * B_, H_, H_, 0);
    wmma_gemm_kernel<<<g, 256, 0, stream>>>(seq, pWih0 + (size_t)H_ * H_, comb0 + H_,
                                            Xpre + (size_t)T_ * B_ * H_,
                                            T_ * B_, H_, H_, 0);
  }
  rnn_scan_kernel<<<2, 1024, 2 * B_ * H_ * sizeof(__bf16), stream>>>(Xpre, pWhh0, exp_b, y0);

  // Layer 1: input GEMM over concat (K=2048), then scan
  {
    dim3 g(T_ * B_ / 128, H_ / 128);
    wmma_gemm_kernel<<<g, 256, 0, stream>>>(y0, pWih1, comb1, Xpre,
                                            T_ * B_, H_, 2 * H_, 0);
    wmma_gemm_kernel<<<g, 256, 0, stream>>>(y0, pWih1 + (size_t)H_ * 2 * H_, comb1 + H_,
                                            Xpre + (size_t)T_ * B_ * H_,
                                            T_ * B_, H_, 2 * H_, 0);
  }
  rnn_scan_kernel<<<2, 1024, 2 * B_ * H_ * sizeof(__bf16), stream>>>(Xpre, pWhh1, exp_b, y1);

  // Projection head
  {
    dim3 g(T_ * B_ / 128, E_ / 128);
    wmma_gemm_kernel<<<g, 256, 0, stream>>>(y1, pWp1, bp1, hp_f, T_ * B_, E_, 2 * H_, 0);
  }
  ln_relu_kernel<<<T_ * B_, 256, 0, stream>>>(hp_f, g2, be2, nullptr, hp_b, E_);

  // Vocab GEMM (134 GFLOP, 524 MB out) with [t,b] -> [b,t] row remap
  {
    dim3 g(T_ * B_ / 128, V_ / 128);
    wmma_gemm_kernel<<<g, 256, 0, stream>>>(hp_b, pWp2, bp2, logits, T_ * B_, V_, E_, 1);
  }
}
